// Decoder_51582557225665
// MI455X (gfx1250) — matrix operand order repaired and checked by reading, compiled, not run on hardware
//
#include <hip/hip_runtime.h>

// ---- model constants ----
#define LL     2
#define SS     2048
#define CACHE_ 2048
#define TT     4096      // CACHE + S
#define HH     16
#define HKV_   8
#define DD     128
#define DMM    2048
#define FFF    8192
#define VV     32000

typedef __attribute__((ext_vector_type(16))) __bf16 v16bf;
typedef __attribute__((ext_vector_type(8)))  float  v8f;
typedef __attribute__((ext_vector_type(4)))  unsigned int v4u;
typedef __attribute__((ext_vector_type(4)))  int v4i;
typedef __attribute__((ext_vector_type(8)))  int v8i;

#if defined(__has_builtin)
#  if __has_builtin(__builtin_amdgcn_tensor_load_to_lds)
#    define HAVE_TDM 1
#  else
#    define HAVE_TDM 0
#  endif
#else
#  define HAVE_TDM 0
#endif

union Frag {
  v16bf v;
  unsigned short u[16];
  unsigned int   d[8];
};

__device__ __forceinline__ unsigned short f2bf(float f) {
  unsigned int b = __float_as_uint(f);
  b += 0x7FFFu + ((b >> 16) & 1u);           // round-to-nearest-even
  return (unsigned short)(b >> 16);
}
__device__ __forceinline__ int a_koff(int e, int hf) {   // A-frag 16x32 bf16 K offset
  return 2 * e + 8 * hf + (e >= 4 ? 8 : 0);
}

// ---- TDM: load one 128-row x 32-col bf16 tile (row stride = stride0 elems) into LDS ----
__device__ __forceinline__ void tdm_load_tile(unsigned lds_off, const unsigned short* gsrc,
                                              int tensor_d0, int tensor_d1, long long stride0) {
#if HAVE_TDM
  unsigned long long ga = (unsigned long long)gsrc;
  v4u g0;
  g0[0] = 1u;                                   // count=1, user descriptor
  g0[1] = lds_off;                              // lds_addr (bytes)
  g0[2] = (unsigned)ga;                         // global_addr[31:0]
  g0[3] = (unsigned)(ga >> 32) | (2u << 30);    // global_addr[56:32] | type=2 ("image")
  v8i g1;
  g1[0] = (int)(1u << 16);                      // data_size=1 -> 2-byte elements
  g1[1] = (int)(((unsigned)tensor_d0 & 0xFFFFu) << 16);                         // tensor_dim0[15:0]
  g1[2] = (int)(((unsigned)tensor_d0 >> 16) | (((unsigned)tensor_d1 & 0xFFFFu) << 16));
  g1[3] = (int)(((unsigned)tensor_d1 >> 16) | (32u << 16));                     // tile_dim0 = 32
  g1[4] = 128;                                  // tile_dim1 = 128, tile_dim2 = 0
  g1[5] = (int)(unsigned)((unsigned long long)stride0 & 0xFFFFFFFFull);         // dim0 stride lo
  g1[6] = (int)(unsigned)((unsigned long long)stride0 >> 32);                   // dim0 stride hi
  g1[7] = 0;
  v4i z4 = {0, 0, 0, 0};
#if __clang_major__ >= 23
  v8i z8 = {0, 0, 0, 0, 0, 0, 0, 0};
  __builtin_amdgcn_tensor_load_to_lds(g0, g1, z4, z4, z8, 0);
#else
  __builtin_amdgcn_tensor_load_to_lds(g0, g1, z4, z4, 0);
#endif
#endif
}

// ---------------- weight convert + transpose: W[K][N] f32 -> Wt[N][K] bf16 ----------------
__global__ __launch_bounds__(256)
void convert_wt_kernel(const float* __restrict__ W, unsigned short* __restrict__ Wt,
                       int K, int N) {
  __shared__ float tile[32][33];
  int bk = blockIdx.x * 32, bn = blockIdx.y * 32;
  int tx = threadIdx.x & 31, ty = threadIdx.x >> 5;
  for (int r = ty; r < 32; r += 8)
    tile[r][tx] = W[(size_t)(bk + r) * N + bn + tx];
  __syncthreads();
  for (int r = ty; r < 32; r += 8)
    Wt[(size_t)(bn + r) * K + bk + tx] = f2bf(tile[tx][r]);
}

// ---------------- embedding gather ----------------
__global__ __launch_bounds__(256)
void embed_kernel(const int* __restrict__ ids, const float* __restrict__ emb,
                  float* __restrict__ h) {
  int s = blockIdx.x;
  const float* src = emb + (size_t)ids[s] * DMM;
  float* dst = h + (size_t)s * DMM;
  for (int c = threadIdx.x; c < DMM; c += 256) dst[c] = src[c];
}

// ---------------- RMSNorm (f32 in, bf16 out) ----------------
__global__ __launch_bounds__(256)
void rmsnorm_bf_kernel(const float* __restrict__ in, const float* __restrict__ w,
                       unsigned short* __restrict__ out, int ncols) {
  int row = blockIdx.x;
  const float* x = in + (size_t)row * ncols;
  float ss = 0.f;
  for (int c = threadIdx.x; c < ncols; c += 256) { float v = x[c]; ss += v * v; }
  for (int o = 16; o > 0; o >>= 1) ss += __shfl_xor(ss, o, 32);
  __shared__ float red[8];
  if ((threadIdx.x & 31) == 0) red[threadIdx.x >> 5] = ss;
  __syncthreads();
  float tot = 0.f;
#pragma unroll
  for (int i = 0; i < 8; ++i) tot += red[i];
  float inv = rsqrtf(tot / (float)ncols + 1e-6f);
  for (int c = threadIdx.x; c < ncols; c += 256)
    out[(size_t)row * ncols + c] = f2bf(x[c] * inv * w[c]);
}

// ---- bf16 WMMA GEMM: C[M,N](f32) = A[M,K](bf16) @ Bt[N,K](bf16)^T (+ addsrc), TDM-fed ----
__global__ __launch_bounds__(256)
void gemm_bf_kernel(const unsigned short* __restrict__ A, const unsigned short* __restrict__ Bt,
                    const float* __restrict__ addsrc, float* __restrict__ C,
                    int M, int N, int K) {
#if HAVE_TDM
  __shared__ unsigned short As[2][128][32];   // double-buffered, 16KB
  __shared__ unsigned short Bs[2][128][32];   // 16KB
#else
  __shared__ unsigned short As[1][128][32];
  __shared__ unsigned short Bs[1][128][32];
#endif
  int tid = threadIdx.x;
  int lane = tid & 31, wave = tid >> 5;
  int hf = lane >> 4, l16 = lane & 15;
  int wm = wave >> 1, wn = wave & 1;          // 4x2 wave grid -> 32x64 tile per wave
  int bm = blockIdx.y * 128, bn = blockIdx.x * 128;

  v8f vz = {0.f,0.f,0.f,0.f,0.f,0.f,0.f,0.f};
  v8f acc[2][4];
#pragma unroll
  for (int i = 0; i < 2; ++i)
#pragma unroll
    for (int j = 0; j < 4; ++j) acc[i][j] = vz;

#if HAVE_TDM
  if (wave == 0) {   // prologue: DMA first tiles
    tdm_load_tile((unsigned)(unsigned long long)&As[0][0][0], A  + (size_t)bm * K, K, M, K);
    tdm_load_tile((unsigned)(unsigned long long)&Bs[0][0][0], Bt + (size_t)bn * K, K, N, K);
  }
#endif

  for (int k0 = 0; k0 < K; k0 += 32) {
#if HAVE_TDM
    int cur = (k0 >> 5) & 1;
    if (wave == 0) {
      if (k0 + 32 < K) {   // issue next tiles, then wait for current pair (in-order TENSORcnt)
        tdm_load_tile((unsigned)(unsigned long long)&As[cur ^ 1][0][0],
                      A + (size_t)bm * K + (k0 + 32), K, M, K);
        tdm_load_tile((unsigned)(unsigned long long)&Bs[cur ^ 1][0][0],
                      Bt + (size_t)bn * K + (k0 + 32), K, N, K);
        __builtin_amdgcn_s_wait_tensorcnt(2);
      } else {
        __builtin_amdgcn_s_wait_tensorcnt(0);
      }
    }
    __syncthreads();
#else
    int cur = 0;
#pragma unroll
    for (int j = 0; j < 16; ++j) {
      int idx = tid + 256 * j;
      int r = idx >> 5, c = idx & 31;
      As[0][r][c] = A[(size_t)(bm + r) * K + k0 + c];
      Bs[0][r][c] = Bt[(size_t)(bn + r) * K + k0 + c];
    }
    __syncthreads();
#endif

    Frag af[2], bfr[4];
#pragma unroll
    for (int i = 0; i < 2; ++i) {
      int row = wm * 32 + 16 * i + l16;
      const unsigned int* ar = (const unsigned int*)&As[cur][row][0];
#pragma unroll
      for (int e = 0; e < 8; ++e) af[i].d[e] = ar[a_koff(e, hf) >> 1];
    }
#pragma unroll
    for (int j = 0; j < 4; ++j) {
      int row = wn * 64 + 16 * j + l16;
      const unsigned int* br = (const unsigned int*)&Bs[cur][row][0];
#pragma unroll
      for (int e = 0; e < 8; ++e) bfr[j].d[e] = br[a_koff(e, hf) >> 1];
    }
#pragma unroll
    for (int i = 0; i < 2; ++i)
#pragma unroll
      for (int j = 0; j < 4; ++j)
        acc[i][j] = __builtin_amdgcn_wmma_f32_16x16x32_bf16(
            false, af[i].v, false, bfr[j].v, (short)0, acc[i][j], false, false);
    __syncthreads();
  }

#pragma unroll
  for (int i = 0; i < 2; ++i)
#pragma unroll
    for (int j = 0; j < 4; ++j) {
      int col = bn + wn * 64 + 16 * j + l16;
#pragma unroll
      for (int r = 0; r < 8; ++r) {
        int row = bm + wm * 32 + 16 * i + r + 8 * hf;
        size_t idx = (size_t)row * N + col;
        float v = acc[i][j][r];
        if (addsrc) v += addsrc[idx];
        C[idx] = v;
      }
    }
}

// ---------------- RoPE on q (in place), pos = CACHE + s ----------------
__global__ __launch_bounds__(256)
void rope_q_kernel(float* __restrict__ q) {
  int s = blockIdx.x;
  float pos = (float)(CACHE_ + s);
  float* row = q + (size_t)s * (HH * DD);
  for (int p = threadIdx.x; p < HH * 64; p += 256) {
    int hh = p >> 6, i = p & 63;
    float freq = __powf(10000.0f, -(float)i * (1.0f / 64.0f));
    float sn, cs; __sincosf(pos * freq, &sn, &cs);
    float x0 = row[hh * DD + i], x1 = row[hh * DD + 64 + i];
    row[hh * DD + i]      = x0 * cs - x1 * sn;
    row[hh * DD + 64 + i] = x1 * cs + x0 * sn;
  }
}

// ---------------- build full K (roped) / V from cache + new GQA kv ----------------
__global__ __launch_bounds__(64)
void build_kv_kernel(const float* __restrict__ kc, const float* __restrict__ vc,
                     const float* __restrict__ kn, const float* __restrict__ vn,
                     float* __restrict__ Kf, float* __restrict__ Vf) {
  int t = blockIdx.x, hh = blockIdx.y, i = threadIdx.x;  // i < 64
  const float *ksrc, *vsrc;
  if (t < CACHE_) {
    ksrc = kc + ((size_t)hh * CACHE_ + t) * DD;
    vsrc = vc + ((size_t)hh * CACHE_ + t) * DD;
  } else {
    int s = t - CACHE_;
    ksrc = kn + ((size_t)s * HKV_ + (hh >> 1)) * DD;   // repeat_kv: head h -> kv head h/2
    vsrc = vn + ((size_t)s * HKV_ + (hh >> 1)) * DD;
  }
  float freq = __powf(10000.0f, -(float)i * (1.0f / 64.0f));
  float sn, cs; __sincosf((float)t * freq, &sn, &cs);
  float x0 = ksrc[i], x1 = ksrc[64 + i];
  size_t o = ((size_t)hh * TT + t) * DD;
  Kf[o + i]      = x0 * cs - x1 * sn;
  Kf[o + 64 + i] = x1 * cs + x0 * sn;
  Vf[o + i]      = vsrc[i];
  Vf[o + 64 + i] = vsrc[64 + i];
}

// ---------------- flash attention: per (head, 64-query block), 4 waves, bf16 out ----------------
__global__ __launch_bounds__(128)
void attn_kernel(const float* __restrict__ Q, const float* __restrict__ Kf,
                 const float* __restrict__ Vf, unsigned short* __restrict__ O) {
  __shared__ unsigned short pbuf[4][16][32];   // per-wave P tile (bf16)
  int head = blockIdx.x;
  int wave = threadIdx.x >> 5, lane = threadIdx.x & 31;
  int hf = lane >> 4, l16 = lane & 15;
  int qbase = blockIdx.y * 64 + wave * 16;
  const size_t hoff = (size_t)head * TT * DD;
  const float scale = 0.08838834764831843f;    // 1/sqrt(128)

  Frag qa[4];
  const float* qrow = Q + (size_t)(qbase + l16) * (HH * DD) + head * DD;
#pragma unroll
  for (int c = 0; c < 4; ++c)
#pragma unroll
    for (int e = 0; e < 8; ++e) {
      int ka = a_koff(e, hf);
      qa[c].u[2 * e]     = f2bf(qrow[c * 32 + ka]);
      qa[c].u[2 * e + 1] = f2bf(qrow[c * 32 + ka + 1]);
    }

  v8f vz = {0.f,0.f,0.f,0.f,0.f,0.f,0.f,0.f};
  v8f o_acc[8];
#pragma unroll
  for (int n = 0; n < 8; ++n) o_acc[n] = vz;
  float mrow[8], lrow[8];
#pragma unroll
  for (int r = 0; r < 8; ++r) { mrow[r] = -3.0e38f; lrow[r] = 0.f; }

  int ktend = CACHE_ + blockIdx.y * 64 + 64;   // uniform over block -> barriers legal

  for (int kt = 0; kt < ktend; kt += 32) {
    v8f s0 = vz, s1 = vz;
#pragma unroll
    for (int c = 0; c < 4; ++c) {              // scores = q(16xD) @ K^T, D in chunks of 32
      Frag bk0, bk1;
      const float* kr0 = Kf + hoff + (size_t)(kt + l16) * DD + c * 32;
      const float* kr1 = kr0 + 16 * DD;
#pragma unroll
      for (int e = 0; e < 8; ++e) {
        int kb = a_koff(e, hf);
        bk0.u[2 * e] = f2bf(kr0[kb]); bk0.u[2 * e + 1] = f2bf(kr0[kb + 1]);
        bk1.u[2 * e] = f2bf(kr1[kb]); bk1.u[2 * e + 1] = f2bf(kr1[kb + 1]);
      }
      s0 = __builtin_amdgcn_wmma_f32_16x16x32_bf16(false, qa[c].v, false, bk0.v, (short)0, s0, false, false);
      s1 = __builtin_amdgcn_wmma_f32_16x16x32_bf16(false, qa[c].v, false, bk1.v, (short)0, s1, false, false);
    }
    float p0[8], p1[8];
#pragma unroll
    for (int r = 0; r < 8; ++r) {
      int qpos = CACHE_ + qbase + r + 8 * hf;
      float a = s0[r] * scale; if (kt + l16 > qpos)      a = -1e30f;
      float b = s1[r] * scale; if (kt + 16 + l16 > qpos) b = -1e30f;
      float cm = fmaxf(a, b);
      cm = fmaxf(cm, __shfl_xor(cm, 1, 32));
      cm = fmaxf(cm, __shfl_xor(cm, 2, 32));
      cm = fmaxf(cm, __shfl_xor(cm, 4, 32));
      cm = fmaxf(cm, __shfl_xor(cm, 8, 32));
      float mnew = fmaxf(mrow[r], cm);
      float fac = __expf(mrow[r] - mnew);
      float e0 = __expf(a - mnew), e1 = __expf(b - mnew);
      float rs = e0 + e1;
      rs += __shfl_xor(rs, 1, 32); rs += __shfl_xor(rs, 2, 32);
      rs += __shfl_xor(rs, 4, 32); rs += __shfl_xor(rs, 8, 32);
      lrow[r] = lrow[r] * fac + rs;
      mrow[r] = mnew;
      p0[r] = e0; p1[r] = e1;
#pragma unroll
      for (int n = 0; n < 8; ++n) o_acc[n][r] *= fac;
    }
#pragma unroll
    for (int r = 0; r < 8; ++r) {              // C-layout -> A-fragment transpose via LDS
      pbuf[wave][r + 8 * hf][l16]      = f2bf(p0[r]);
      pbuf[wave][r + 8 * hf][16 + l16] = f2bf(p1[r]);
    }
    __syncthreads();
    Frag pa;
#pragma unroll
    for (int e = 0; e < 8; ++e) {
      int ka = a_koff(e, hf);
      pa.u[2 * e]     = pbuf[wave][l16][ka];
      pa.u[2 * e + 1] = pbuf[wave][l16][ka + 1];
    }
#pragma unroll
    for (int n = 0; n < 8; ++n) {              // O += P(16x32) @ V(32x128)
      Frag vb;
      const float* vcol = Vf + hoff + (size_t)kt * DD + n * 16 + l16;
#pragma unroll
      for (int e = 0; e < 8; ++e) {
        int kb = a_koff(e, hf);
        vb.u[2 * e]     = f2bf(vcol[(size_t)kb * DD]);
        vb.u[2 * e + 1] = f2bf(vcol[(size_t)(kb + 1) * DD]);
      }
      o_acc[n] = __builtin_amdgcn_wmma_f32_16x16x32_bf16(false, pa.v, false, vb.v, (short)0, o_acc[n], false, false);
    }
    __syncthreads();
  }

#pragma unroll
  for (int r = 0; r < 8; ++r) {
    float inv = 1.0f / lrow[r];
    int m = r + 8 * hf;
    unsigned short* orow = O + (size_t)(qbase + m) * (HH * DD) + head * DD + l16;
#pragma unroll
    for (int n = 0; n < 8; ++n) orow[n * 16] = f2bf(o_acc[n][r] * inv);
  }
}

// ---------------- SiLU(g) * u -> bf16 ----------------
__global__ __launch_bounds__(256)
void silu_mul_bf_kernel(const float* __restrict__ g, const float* __restrict__ u,
                        unsigned short* __restrict__ gb, int n) {
  for (int i = blockIdx.x * blockDim.x + threadIdx.x; i < n; i += gridDim.x * blockDim.x) {
    float x = g[i];
    gb[i] = f2bf(x / (1.0f + __expf(-x)) * u[i]);
  }
}

// ---------------- host launch ----------------
extern "C" void kernel_launch(void* const* d_in, const int* in_sizes, int n_in,
                              void* d_out, int out_size, void* d_ws, size_t ws_size,
                              hipStream_t stream) {
  const int*   ids     = (const int*)d_in[0];
  const float* embed   = (const float*)d_in[1];
  const float* Wq      = (const float*)d_in[2];
  const float* Wk      = (const float*)d_in[3];
  const float* Wv      = (const float*)d_in[4];
  const float* Wo      = (const float*)d_in[5];
  const float* Wg      = (const float*)d_in[6];
  const float* Wu      = (const float*)d_in[7];
  const float* Wd      = (const float*)d_in[8];
  const float* ln1     = (const float*)d_in[9];
  const float* ln2     = (const float*)d_in[10];
  const float* fnorm   = (const float*)d_in[11];
  const float* lm_head = (const float*)d_in[12];
  const float* k_cache = (const float*)d_in[13];
  const float* v_cache = (const float*)d_in[14];
  float* out = (float*)d_out;

  float* ws = (float*)d_ws;
  size_t off = 0;
  auto alloc = [&](size_t nfloats) { float* p = ws + off; off += nfloats; return p; };
  float*          h   = alloc((size_t)SS * DMM);
  unsigned short* xb  = (unsigned short*)alloc((size_t)SS * DMM / 2);
  float*          q   = alloc((size_t)SS * HH * DD);
  float*          kn  = alloc((size_t)SS * HKV_ * DD);
  float*          vn  = alloc((size_t)SS * HKV_ * DD);
  float*          Kf  = alloc((size_t)HH * TT * DD);
  float*          Vf  = alloc((size_t)HH * TT * DD);
  unsigned short* aob = (unsigned short*)alloc((size_t)SS * HH * DD / 2);
  float*          g   = alloc((size_t)SS * FFF);
  float*          u   = alloc((size_t)SS * FFF);
  unsigned short* gb  = (unsigned short*)alloc((size_t)SS * FFF / 2);
  unsigned short* wt  = (unsigned short*)alloc((size_t)DMM * VV / 2);  // bf16 weight arena (largest)

  auto cvtw = [&](const float* W, int K, int N) {   // W[K][N] f32 -> wt[N][K] bf16
    convert_wt_kernel<<<dim3(K / 32, N / 32), 256, 0, stream>>>(W, wt, K, N);
  };

  embed_kernel<<<SS, 256, 0, stream>>>(ids, embed, h);

  for (int l = 0; l < LL; ++l) {
    const float* wq = Wq + (size_t)l * DMM * (HH * DD);
    const float* wk = Wk + (size_t)l * DMM * (HKV_ * DD);
    const float* wv = Wv + (size_t)l * DMM * (HKV_ * DD);
    const float* wo = Wo + (size_t)l * (HH * DD) * DMM;
    const float* wg = Wg + (size_t)l * DMM * FFF;
    const float* wu = Wu + (size_t)l * DMM * FFF;
    const float* wd = Wd + (size_t)l * FFF * DMM;
    const float* kc = k_cache + (size_t)l * HH * CACHE_ * DD;
    const float* vc = v_cache + (size_t)l * HH * CACHE_ * DD;

    rmsnorm_bf_kernel<<<SS, 256, 0, stream>>>(h, ln1 + (size_t)l * DMM, xb, DMM);
    cvtw(wq, DMM, HH * DD);
    gemm_bf_kernel<<<dim3((HH * DD) / 128, SS / 128), 256, 0, stream>>>(xb, wt, nullptr, q, SS, HH * DD, DMM);
    cvtw(wk, DMM, HKV_ * DD);
    gemm_bf_kernel<<<dim3((HKV_ * DD) / 128, SS / 128), 256, 0, stream>>>(xb, wt, nullptr, kn, SS, HKV_ * DD, DMM);
    cvtw(wv, DMM, HKV_ * DD);
    gemm_bf_kernel<<<dim3((HKV_ * DD) / 128, SS / 128), 256, 0, stream>>>(xb, wt, nullptr, vn, SS, HKV_ * DD, DMM);
    rope_q_kernel<<<SS, 256, 0, stream>>>(q);
    build_kv_kernel<<<dim3(TT, HH), 64, 0, stream>>>(kc, vc, kn, vn, Kf, Vf);
    attn_kernel<<<dim3(HH, SS / 64), 128, 0, stream>>>(q, Kf, Vf, aob);
    cvtw(wo, HH * DD, DMM);
    gemm_bf_kernel<<<dim3(DMM / 128, SS / 128), 256, 0, stream>>>(aob, wt, h, h, SS, DMM, HH * DD);

    rmsnorm_bf_kernel<<<SS, 256, 0, stream>>>(h, ln2 + (size_t)l * DMM, xb, DMM);
    cvtw(wg, DMM, FFF);
    gemm_bf_kernel<<<dim3(FFF / 128, SS / 128), 256, 0, stream>>>(xb, wt, nullptr, g, SS, FFF, DMM);
    cvtw(wu, DMM, FFF);
    gemm_bf_kernel<<<dim3(FFF / 128, SS / 128), 256, 0, stream>>>(xb, wt, nullptr, u, SS, FFF, DMM);
    silu_mul_bf_kernel<<<4096, 256, 0, stream>>>(g, u, gb, SS * FFF);
    cvtw(wd, FFF, DMM);
    gemm_bf_kernel<<<dim3(DMM / 128, SS / 128), 256, 0, stream>>>(gb, wt, h, h, SS, DMM, FFF);
  }

  rmsnorm_bf_kernel<<<SS, 256, 0, stream>>>(h, fnorm, xb, DMM);
  cvtw(lm_head, DMM, VV);
  gemm_bf_kernel<<<dim3(VV / 128, SS / 128), 256, 0, stream>>>(xb, wt, nullptr, out, SS, VV, DMM);
}
